// DeformableConv2D_80461917323527
// MI455X (gfx1250) — compile-verified, matrix operand order checked by reading, not run on hardware
//
#include <hip/hip_runtime.h>
#include <hip/hip_bf16.h>

typedef _Float16 v16h __attribute__((ext_vector_type(16)));
typedef _Float16 v8h  __attribute__((ext_vector_type(8)));
typedef float    v8f  __attribute__((ext_vector_type(8)));

#define Bc   8
#define Hc   128
#define Wc   128
#define CINc 64
#define GCc  32      // channels per group
#define SCc  288     // 9 * 32
#define GFc  64      // output channels per group
#define TIL  8       // spatial tile (8x8 output pixels -> M = 64)

__global__ __launch_bounds__(256, 1)
void deform_conv_fused(const float* __restrict__ x,
                       const float* __restrict__ off_w,
                       const float* __restrict__ off_b,
                       const float* __restrict__ dw_w,
                       const float* __restrict__ dw_b,
                       const float* __restrict__ pw_w,
                       const float* __restrict__ pw_b,
                       float* __restrict__ out)
{
    // ---- LDS plan (~190 KB, fits CDNA5's 320 KB WGP LDS) ----
    __shared__ float    s_x[144 * 32];        // 12x12 x-halo tile, 32 ch (18.4 KB)
    __shared__ float    s_offw[5184];         // 3*3*32*18 offset-conv weights (20.7 KB)
    __shared__ float    s_dww[2592];          // 3*3*288 depthwise weights (10.4 KB)
    __shared__ _Float16 s_pwT[GFc * SCc];     // pointwise weights, N-major [n][k], f16 (36.9 KB)
    __shared__ float    s_off[100 * 18];      // offsets for 10x10 samp-halo (7.2 KB)
    __shared__ _Float16 s_samp[100 * SCc];    // bilinear samples, f16 (57.6 KB)
    __shared__ _Float16 s_y[64 * SCc];        // depthwise output [pix][k], f16 (36.9 KB)
    __shared__ float    s_offb[18];
    __shared__ float    s_dwb[SCc];
    __shared__ float    s_pwb[GFc];

    const int tid = threadIdx.x;
    const int tx  = blockIdx.x & 15;
    const int ty  = blockIdx.x >> 4;
    const int bb  = blockIdx.y;
    const int g   = blockIdx.z;

    // ---- P0: stage weights/biases in LDS (pw transposed to N-major) ----
    for (int i = tid; i < 5184; i += 256)      s_offw[i] = off_w[g * 5184 + i];
    for (int i = tid; i < 2592; i += 256)      s_dww[i]  = dw_w[g * 2592 + i];
    for (int i = tid; i < SCc * GFc; i += 256) {
        int k = i >> 6, n = i & 63;            // coalesced global read (i = k*64 + n)
        s_pwT[n * SCc + k] = (_Float16)pw_w[g * SCc * GFc + i];
    }
    if (tid < 18)                              s_offb[tid] = off_b[g * 18 + tid];
    for (int i = tid; i < SCc; i += 256)       s_dwb[i]  = dw_b[g * SCc + i];
    if (tid < GFc)                             s_pwb[tid] = pw_b[g * GFc + tid];

    // ---- P1: load 12x12 x tile (halo +-2), zero-pad outside image (SAME) ----
    for (int i = tid; i < 144 * 32; i += 256) {
        int p = i >> 5, c = i & 31;
        int ry = p / 12 - 2, rx = p % 12 - 2;
        int gy = ty * TIL + ry, gx = tx * TIL + rx;
        float v = 0.0f;
        if ((unsigned)gy < (unsigned)Hc && (unsigned)gx < (unsigned)Wc)
            v = x[(((size_t)bb * Hc + gy) * Wc + gx) * CINc + g * GCc + c];
        s_x[p * 32 + c] = v;
    }
    __syncthreads();

    // ---- P2: offset conv (3x3x32 -> 18) for 10x10 samp-halo pixels ----
    for (int i = tid; i < 100 * 18; i += 256) {
        int p = i / 18, o = i - p * 18;
        int sy = p / 10 - 1, sx = p % 10 - 1;
        int gy = ty * TIL + sy, gx = tx * TIL + sx;
        float acc = 0.0f;
        if ((unsigned)gy < (unsigned)Hc && (unsigned)gx < (unsigned)Wc) {
            acc = s_offb[o];
            for (int k = 0; k < 9; ++k) {
                int ky = k / 3, kx = k % 3;
                const float* xp = &s_x[((sy + ky + 1) * 12 + (sx + kx + 1)) * 32];
                const float* wp = &s_offw[k * 32 * 18 + o];
                #pragma unroll
                for (int c = 0; c < 32; ++c) acc += xp[c] * wp[c * 18];
            }
        }
        s_off[p * 18 + o] = acc;
    }
    __syncthreads();

    // ---- P3: bilinear deformable sampling -> s_samp (f16) ----
    for (int i = tid; i < 900; i += 256) {
        int p = i / 9, kk = i - p * 9;
        int sy = p / 10 - 1, sx = p % 10 - 1;
        int gy = ty * TIL + sy, gx = tx * TIL + sx;
        _Float16* dst = &s_samp[p * SCc + kk * GCc];
        if (!((unsigned)gy < (unsigned)Hc && (unsigned)gx < (unsigned)Wc)) {
            #pragma unroll
            for (int c = 0; c < GCc; ++c) dst[c] = (_Float16)0.0f;
            continue;
        }
        float ox = s_off[p * 18 + kk * 2 + 0];
        float oy = s_off[p * 18 + kk * 2 + 1];
        float lx = fminf(fmaxf((float)gx + (float)(kk % 3 - 1) + ox, 0.0f), (float)(Wc - 1));
        float ly = fminf(fmaxf((float)gy + (float)(kk / 3 - 1) + oy, 0.0f), (float)(Hc - 1));
        float x0f = floorf(lx), y0f = floorf(ly);
        float fx0 = fminf(fmaxf(x0f,        0.0f), (float)(Wc - 1));
        float fx1 = fminf(fmaxf(x0f + 1.0f, 0.0f), (float)(Wc - 1));
        float fy0 = fminf(fmaxf(y0f,        0.0f), (float)(Hc - 1));
        float fy1 = fminf(fmaxf(y0f + 1.0f, 0.0f), (float)(Hc - 1));
        int x0 = (int)fx0, x1 = (int)fx1, y0 = (int)fy0, y1 = (int)fy1;
        float wa  = (fx1 - lx) * (fy1 - ly);
        float wb  = (fx1 - lx) * (ly - fy0);
        float wcw = (lx - fx0) * (fy1 - ly);
        float wd  = (lx - fx0) * (ly - fy0);
        const float4* pa = (const float4*)&x[(((size_t)bb * Hc + y0) * Wc + x0) * CINc + g * GCc];
        const float4* pb = (const float4*)&x[(((size_t)bb * Hc + y1) * Wc + x0) * CINc + g * GCc];
        const float4* pc = (const float4*)&x[(((size_t)bb * Hc + y0) * Wc + x1) * CINc + g * GCc];
        const float4* pd = (const float4*)&x[(((size_t)bb * Hc + y1) * Wc + x1) * CINc + g * GCc];
        #pragma unroll
        for (int q = 0; q < 8; ++q) {
            float4 A = pa[q], Bv = pb[q], Cv = pc[q], D = pd[q];
            dst[q * 4 + 0] = (_Float16)(wa * A.x + wb * Bv.x + wcw * Cv.x + wd * D.x);
            dst[q * 4 + 1] = (_Float16)(wa * A.y + wb * Bv.y + wcw * Cv.y + wd * D.y);
            dst[q * 4 + 2] = (_Float16)(wa * A.z + wb * Bv.z + wcw * Cv.z + wd * D.z);
            dst[q * 4 + 3] = (_Float16)(wa * A.w + wb * Bv.w + wcw * Cv.w + wd * D.w);
        }
    }
    __syncthreads();

    // ---- P4: depthwise 3x3 over 288 channels -> s_y (f16) ----
    for (int i = tid; i < 64 * SCc; i += 256) {
        int p = i / SCc, ch = i - p * SCc;
        int lyy = p >> 3, lxx = p & 7;
        float acc = s_dwb[ch];
        #pragma unroll
        for (int k = 0; k < 9; ++k) {
            int ny = lyy + 1 + k / 3 - 1;
            int nx = lxx + 1 + k % 3 - 1;
            acc += s_dww[k * SCc + ch] * (float)s_samp[(ny * 10 + nx) * SCc + ch];
        }
        s_y[p * SCc + ch] = (_Float16)acc;
    }
    __syncthreads();

    // ---- P5: pointwise conv as WMMA GEMM: (64 x 288) @ (288 x 64) ----
    // 16-bit A-fragment layout (ISA 7.12.2): lane<16 -> M=lane,   K in {0..7, 16..23}
    //                                        lane>=16 -> M=lane-16, K in {8..15, 24..31}
    // => per lane two contiguous 8-half runs => two ds_load_b128 per fragment.
    const int wv   = tid >> 5;
    const int lane = tid & 31;
    const int Mblock = wv & 3;            // 4 M-blocks of 16 pixels
    const int nb0    = (wv >> 2) * 2;     // 2 N-blocks of 16 cols per wave
    const int mloc   = lane & 15;
    const int pixA   = Mblock * 16 + mloc;
    const int kb     = (lane < 16) ? 0 : 8;
    const int n0     = nb0 * 16 + mloc;

    const _Float16* yrow  = &s_y[pixA * SCc];
    const _Float16* b0row = &s_pwT[n0 * SCc];
    const _Float16* b1row = &s_pwT[(n0 + 16) * SCc];

    v8f acc0 = {};
    v8f acc1 = {};
    #pragma unroll
    for (int kc = 0; kc < 9; ++kc) {
        const int base = kc * 32 + kb;           // 16B-aligned half offset
        v8h alo  = *(const v8h*)(yrow  + base);
        v8h ahi  = *(const v8h*)(yrow  + base + 16);
        v8h b0lo = *(const v8h*)(b0row + base);
        v8h b0hi = *(const v8h*)(b0row + base + 16);
        v8h b1lo = *(const v8h*)(b1row + base);
        v8h b1hi = *(const v8h*)(b1row + base + 16);
        v16h a  = __builtin_shufflevector(alo,  ahi,  0,1,2,3,4,5,6,7,8,9,10,11,12,13,14,15);
        v16h b0 = __builtin_shufflevector(b0lo, b0hi, 0,1,2,3,4,5,6,7,8,9,10,11,12,13,14,15);
        v16h b1 = __builtin_shufflevector(b1lo, b1hi, 0,1,2,3,4,5,6,7,8,9,10,11,12,13,14,15);
        acc0 = __builtin_amdgcn_wmma_f32_16x16x32_f16(false, a, false, b0,
                                                      (short)0, acc0, false, false);
        acc1 = __builtin_amdgcn_wmma_f32_16x16x32_f16(false, a, false, b1,
                                                      (short)0, acc1, false, false);
    }

    // ---- store: C/D layout -> VGPR i holds M = i (+8 for lanes 16-31), N = lane&15 ----
    #pragma unroll
    for (int i = 0; i < 8; ++i) {
        int Mi = i + ((lane >= 16) ? 8 : 0);
        int p  = Mblock * 16 + Mi;
        int lyy = p >> 3, lxx = p & 7;
        int gy = ty * TIL + lyy, gx = tx * TIL + lxx;
        float* op = &out[(((size_t)bb * Hc + gy) * Wc + gx) * 128 + g * GFc];
        op[n0]      = acc0[i] + s_pwb[n0];
        op[n0 + 16] = acc1[i] + s_pwb[n0 + 16];
    }
}

extern "C" void kernel_launch(void* const* d_in, const int* in_sizes, int n_in,
                              void* d_out, int out_size, void* d_ws, size_t ws_size,
                              hipStream_t stream) {
    const float* x    = (const float*)d_in[0];
    const float* offw = (const float*)d_in[1];
    const float* offb = (const float*)d_in[2];
    const float* dww  = (const float*)d_in[3];
    const float* dwb  = (const float*)d_in[4];
    const float* pww  = (const float*)d_in[5];
    const float* pwb  = (const float*)d_in[6];
    float* out = (float*)d_out;

    dim3 grid(256, Bc, 2);   // 16x16 spatial tiles of 8x8, batch, group
    dim3 block(256);
    hipLaunchKernelGGL(deform_conv_fused, grid, block, 0, stream,
                       x, offw, offb, dww, dwb, pww, pwb, out);
}
